// LeapfrogIntegrator_3822520893674
// MI455X (gfx1250) — matrix operand order repaired and checked by reading, not compiled, not run on hardware
//
#include <hip/hip_runtime.h>
#include <hip/hip_bf16.h>
#include <math.h>

// Problem constants (match reference)
#define BB 65536
#define DD 256
#define HH 512
#define EFF_DT 0.1f
#define HALF_DT 0.05f

typedef __attribute__((ext_vector_type(16))) __bf16 bf16x16;
typedef __attribute__((ext_vector_type(8)))  __bf16 bf16x8;
typedef __attribute__((ext_vector_type(8)))  float  f32x8;
typedef unsigned int u32;
typedef __attribute__((ext_vector_type(4))) u32 u32x4;
typedef __attribute__((ext_vector_type(4))) int i32x4;
typedef __attribute__((ext_vector_type(8))) int i32x8;

// LDS layout (dynamic): [ 2x64KB weight staging | hidden | xnew | vhalf ]
#define WBUF_OFF   0
#define HID_OFF    131072
#define XNW_OFF    196608
#define VHF_OFF    229376
#define LDS_BYTES  294912

// ---------------------------------------------------------------------------
// tanh: CDNA5 native V_TANH (single TRANS op) with branchless fallback.
// ---------------------------------------------------------------------------
__device__ __forceinline__ float fast_tanh(float x) {
#if defined(__HIP_DEVICE_COMPILE__)
#if __has_builtin(__builtin_amdgcn_tanhf)
  return __builtin_amdgcn_tanhf(x);
#else
  float ax = __builtin_fabsf(x);
  float e  = __builtin_amdgcn_exp2f(ax * 2.885390082e+00f);
  float t  = 1.0f - 2.0f * __builtin_amdgcn_rcpf(e + 1.0f);
  return __builtin_copysignf(t, x);
#endif
#else
  return tanhf(x);  // host stub, never executed
#endif
}

// ---------------------------------------------------------------------------
// Tensor Data Mover: one 64KB contiguous global->LDS chunk per issue.
// ---------------------------------------------------------------------------
#if defined(__HIP_DEVICE_COMPILE__)
#if __has_builtin(__builtin_amdgcn_tensor_load_to_lds) && \
    __has_builtin(__builtin_amdgcn_s_wait_tensorcnt)
#define HAVE_TDM 1
#endif
#endif

__device__ __forceinline__ void tdm_issue_64k(const void* gsrc, u32 lds_byte_off) {
#ifdef HAVE_TDM
  // D# group 0: count=1 | lds_addr | 57-bit global_addr | type=2
  u32 lds_addr = __builtin_amdgcn_groupstaticsize() + lds_byte_off;
  unsigned long long ga = (unsigned long long)gsrc;
  u32x4 g0;
  g0[0] = 1u;
  g0[1] = lds_addr;
  g0[2] = (u32)ga;
  g0[3] = (u32)((ga >> 32) & 0x1FFFFFFu) | (2u << 30);
  // D# group 1: data_size=8B; tensor 2048x4 (8B units) = 64KB; tile = tensor
  i32x8 g1;
  g1[0] = (3 << 16);     // workgroup_mask=0, data_size=3 (8B)
  g1[1] = (2048 << 16);  // tensor_dim0[15:0] at bits 63:48
  g1[2] = (4 << 16);     // tensor_dim0 hi=0; tensor_dim1[15:0] at bits 111:80
  g1[3] = (2048 << 16);  // tensor_dim1 hi=0; tile_dim0 at bits 127:112
  g1[4] = 4;             // tile_dim1 at bits 143:128; tile_dim2=0
  g1[5] = 2048;          // tensor_dim0_stride (8B units)
  g1[6] = 0;
  g1[7] = 0;
  i32x4 z4 = {0, 0, 0, 0};
  i32x8 z8 = {0, 0, 0, 0, 0, 0, 0, 0};
  __builtin_amdgcn_tensor_load_to_lds(g0, g1, z4, z4, z8, 0);
#else
  (void)gsrc; (void)lds_byte_off;
#endif
}

// Wait for staged chunk: allow 1 outstanding (the prefetched next chunk) or 0.
__device__ __forceinline__ void tdm_wait(bool more_outstanding) {
#ifdef HAVE_TDM
  if (more_outstanding) {
    __builtin_amdgcn_s_wait_tensorcnt(1);
  } else {
    __builtin_amdgcn_s_wait_tensorcnt(0);
  }
#else
  (void)more_outstanding;
#endif
}

// Manual cooperative fallback: 128 threads x 32 x 16B = 64KB
__device__ __forceinline__ void stage_copy_64k(const char* gsrc, char* ldst, int tid) {
  const int4* s = (const int4*)gsrc;
  int4* d = (int4*)ldst;
  #pragma unroll
  for (int i = 0; i < 32; ++i) d[tid + i * 128] = s[tid + i * 128];
}

// ---------------------------------------------------------------------------
// Prep: U[D,H] f32 -> Ut[H,D] bf16 ; W[H,D] f32 -> Wt[D,H] bf16
// ---------------------------------------------------------------------------
__global__ __launch_bounds__(256) void prep_weights_bf16(
    const float* __restrict__ U, const float* __restrict__ W,
    __bf16* __restrict__ Ut, __bf16* __restrict__ Wt) {
  int t = blockIdx.x * 256 + threadIdx.x;      // t in [0, H*D)
  int h = t / DD;
  int d = t % DD;
  Ut[h * DD + d] = (__bf16)U[d * HH + h];      // Ut[h][d]  (GEMM1 B: K=d)
  Wt[d * HH + h] = (__bf16)W[h * DD + d];      // Wt[d][h]  (GEMM2 B: K=h)
}

// ---------------------------------------------------------------------------
// Fused leapfrog. Per WG (4 waves, 64 rows): weights stream L2 -> LDS via TDM
// (double-buffered 64KB chunks, shared by all 4 waves); per wave, 16 rows:
//   hidden = tanh(x @ U)   : wmma bf16, B-frags double-buffered from LDS
//   gamma  = (hidden @ W)*v: wmma bf16, A-frags double-buffered, 4 acc
// ---------------------------------------------------------------------------
__global__ __launch_bounds__(128) void leapfrog_fused(
    const float* __restrict__ x, const float* __restrict__ v,
    const float* __restrict__ force,
    const __bf16* __restrict__ Ut, const __bf16* __restrict__ Wt,
    float* __restrict__ out_x, float* __restrict__ out_v) {
  extern __shared__ char smem[];
  __bf16* sh_hidden = (__bf16*)(smem + HID_OFF);
  __bf16* sh_xnew   = (__bf16*)(smem + XNW_OFF);
  float*  sh_vhalf  = (float*)(smem + VHF_OFF);

  const int wave = threadIdx.x >> 5;
  const int lane = threadIdx.x & 31;
  const int l    = lane & 15;
  const int hlf  = lane >> 4;
  const int rowBase = blockIdx.x * 64 + wave * 16;

  __bf16* hid = sh_hidden + wave * (16 * HH);
  __bf16* xnw = sh_xnew   + wave * (16 * DD);
  float*  vhf = sh_vhalf  + wave * (16 * DD);

  bf16x16 afr[8];  // GEMM1 A fragments, K = 0..255

  for (int pass = 0; pass < 2; ++pass) {
    // ---- Build GEMM1 A fragments. Lane: M=l, K runs at c*32+hlf*16 (+0..15).
    if (pass == 0) {
      const float* xr = x + (size_t)(rowBase + l) * DD + hlf * 16;
      #pragma unroll
      for (int c = 0; c < 8; ++c) {
        const float* s0 = xr + c * 32;
        bf16x16 a;
        #pragma unroll
        for (int j = 0; j < 8; ++j) {
          a[j]     = (__bf16)s0[j];
          a[8 + j] = (__bf16)s0[8 + j];
        }
        afr[c] = a;
      }
    } else {
      const __bf16* xr = xnw + l * DD + hlf * 16;
      #pragma unroll
      for (int c = 0; c < 8; ++c) {
        bf16x8 lo = *(const bf16x8*)(xr + c * 32);
        bf16x8 hi = *(const bf16x8*)(xr + c * 32 + 8);
        bf16x16 a;
        #pragma unroll
        for (int j = 0; j < 8; ++j) { a[j] = lo[j]; a[8 + j] = hi[j]; }
        afr[c] = a;
      }
    }

    // ---- GEMM1: hidden = tanh(A @ U). 4 staged chunks x 8 N-tiles.
#ifdef HAVE_TDM
    if (wave == 0) tdm_issue_64k((const char*)Ut, WBUF_OFF);
#endif
    #pragma unroll 1
    for (int c = 0; c < 4; ++c) {
      char* wb = smem + WBUF_OFF + (c & 1) * 65536;
#ifdef HAVE_TDM
      if (wave == 0) {
        if (c + 1 < 4)
          tdm_issue_64k((const char*)Ut + (c + 1) * 65536,
                        WBUF_OFF + ((c + 1) & 1) * 65536);
        tdm_wait(c + 1 < 4);
      }
      __syncthreads();  // chunk c resident for all waves
#else
      stage_copy_64k((const char*)Ut + c * 65536, wb, threadIdx.x);
      __syncthreads();
#endif
      const __bf16* ub = (const __bf16*)wb;
      #pragma unroll 1
      for (int ntl = 0; ntl < 8; ++ntl) {
        const __bf16* bp = ub + (ntl * 16 + l) * DD + hlf * 16;
        f32x8 acc = {};
        // Software-pipelined: load B-frag c2+1 while WMMA c2 executes.
        bf16x16 b0 = *(const bf16x16*)(bp);
        #pragma unroll
        for (int c2 = 0; c2 < 8; ++c2) {
          bf16x16 bn = b0;
          if (c2 + 1 < 8) bn = *(const bf16x16*)(bp + (c2 + 1) * 32);
          acc = __builtin_amdgcn_wmma_f32_16x16x32_bf16(
              false, afr[c2], false, b0, (short)0, acc, false, false);
          b0 = bn;
        }
        const int nt = c * 8 + ntl;
        #pragma unroll
        for (int r = 0; r < 8; ++r)
          hid[(r + 8 * hlf) * HH + nt * 16 + l] = (__bf16)fast_tanh(acc[r]);
      }
      __syncthreads();  // all waves done reading buf (c&1) before refill
    }

    // ---- GEMM2 + epilogue. 4 staged chunks x 4 N-tiles, 4 accumulators.
#ifdef HAVE_TDM
    if (wave == 0) tdm_issue_64k((const char*)Wt, WBUF_OFF);
#endif
    #pragma unroll 1
    for (int c = 0; c < 4; ++c) {
      char* wb = smem + WBUF_OFF + (c & 1) * 65536;
#ifdef HAVE_TDM
      if (wave == 0) {
        if (c + 1 < 4)
          tdm_issue_64k((const char*)Wt + (c + 1) * 65536,
                        WBUF_OFF + ((c + 1) & 1) * 65536);
        tdm_wait(c + 1 < 4);
      }
      __syncthreads();
#else
      stage_copy_64k((const char*)Wt + c * 65536, wb, threadIdx.x);
      __syncthreads();
#endif
      const __bf16* wbb = (const __bf16*)wb;

      f32x8 acc[4];
      #pragma unroll
      for (int t = 0; t < 4; ++t) { f32x8 z = {}; acc[t] = z; }

      const __bf16* hp0 = hid + l * HH + hlf * 16;
      // Software-pipelined A-fragment across the K loop.
      bf16x8 lo = *(const bf16x8*)(hp0);
      bf16x8 hi = *(const bf16x8*)(hp0 + 8);
      #pragma unroll 1
      for (int kc = 0; kc < 16; ++kc) {
        bf16x8 nlo = lo, nhi = hi;
        if (kc + 1 < 16) {
          nlo = *(const bf16x8*)(hp0 + (kc + 1) * 32);
          nhi = *(const bf16x8*)(hp0 + (kc + 1) * 32 + 8);
        }
        bf16x16 a;
        #pragma unroll
        for (int j = 0; j < 8; ++j) { a[j] = lo[j]; a[8 + j] = hi[j]; }
        #pragma unroll
        for (int t = 0; t < 4; ++t) {
          bf16x16 b = *(const bf16x16*)(wbb + (t * 16 + l) * HH + hlf * 16 + kc * 32);
          acc[t] = __builtin_amdgcn_wmma_f32_16x16x32_bf16(
              false, a, false, b, (short)0, acc[t], false, false);
        }
        lo = nlo; hi = nhi;
      }

      #pragma unroll
      for (int t = 0; t < 4; ++t) {
        const int col = (c * 4 + t) * 16 + l;
        #pragma unroll
        for (int r = 0; r < 8; ++r) {
          const int lr = r + 8 * hlf;
          const size_t gi = (size_t)(rowBase + lr) * DD + col;
          if (pass == 0) {
            float vv = v[gi], ff = force[gi];
            float gamma = acc[t][r] * vv;
            float vh = vv + HALF_DT * (ff - gamma);
            float xn = x[gi] + EFF_DT * vh;
            __builtin_nontemporal_store(xn, &out_x[gi]);   // final output 1
            xnw[lr * DD + col] = (__bf16)xn;                // pass-2 GEMM input
            vhf[lr * DD + col] = vh;                        // pass-2 f32 state
          } else {
            float vh = vhf[lr * DD + col];
            float gamma = acc[t][r] * vh;
            float vn = vh + HALF_DT * (force[gi] - gamma);
            __builtin_nontemporal_store(vn, &out_v[gi]);    // final output 2
          }
        }
      }
      __syncthreads();  // done reading buf (c&1) before refill
    }
  }
}

// ---------------------------------------------------------------------------
extern "C" void kernel_launch(void* const* d_in, const int* in_sizes, int n_in,
                              void* d_out, int out_size, void* d_ws, size_t ws_size,
                              hipStream_t stream) {
  const float* x = (const float*)d_in[0];
  const float* v = (const float*)d_in[1];
  const float* f = (const float*)d_in[2];
  const float* U = (const float*)d_in[3];
  const float* W = (const float*)d_in[4];

  __bf16* Ut = (__bf16*)d_ws;                 // 512*256*2 = 256 KB
  __bf16* Wt = Ut + (size_t)HH * DD;          // 256*512*2 = 256 KB
  float* out_x = (float*)d_out;
  float* out_v = out_x + (size_t)BB * DD;

  prep_weights_bf16<<<dim3((HH * DD) / 256), 256, 0, stream>>>(U, W, Ut, Wt);

  leapfrog_fused<<<dim3(BB / 64), 128, LDS_BYTES, stream>>>(
      x, v, f, Ut, Wt, out_x, out_v);
}
